// NATBlock_2602750182048
// MI455X (gfx1250) — matrix operand order repaired and checked by reading, not compiled, not run on hardware
//
#include <hip/hip_runtime.h>
#include <hip/hip_bf16.h>

// ---------------------------------------------------------------------------
// NAT block for MI455X (gfx1250, wave32). GEMMs run through
// v_wmma_f32_16x16x32_bf16 with f32 accumulation; tiles are staged into LDS
// with global_load_async_to_lds_b128 (ASYNCcnt) and double-buffered so DMA
// overlaps WMMA. Neighborhood attention is a VALU online-softmax kernel; the
// downsample conv is im2col + WMMA GEMM.
// ---------------------------------------------------------------------------

typedef unsigned int   u32;
typedef unsigned short u16;
typedef __attribute__((ext_vector_type(16))) __bf16 v16bf;
typedef __attribute__((ext_vector_type(8)))  float  v8f;
typedef __attribute__((ext_vector_type(4)))  u32    v4u;

union Frag16 { v16bf v; v4u u[2]; };

__device__ __forceinline__ u16 f2bf(float f) {
  u32 u = __float_as_uint(f);
  if ((u & 0x7f800000u) == 0x7f800000u) return (u16)(u >> 16);  // inf/nan
  u32 r = u + 0x7fffu + ((u >> 16) & 1u);                       // RNE
  return (u16)(r >> 16);
}

// ---------------------------------------------------------------------------
// LayerNorm over last dim C (128 or 512). One 128-thread block per row.
// ---------------------------------------------------------------------------
__global__ __launch_bounds__(128)
void ln_kernel(const float* __restrict__ x, const float* __restrict__ g,
               const float* __restrict__ b, float* __restrict__ y, int C) {
  const int row = blockIdx.x, t = threadIdx.x;
  const int per = C >> 7;  // C/128
  const float* xr = x + (size_t)row * C;
  float loc[4];
  float s = 0.f;
  for (int i = 0; i < per; ++i) { loc[i] = xr[t + (i << 7)]; s += loc[i]; }
  __shared__ float red[128];
  red[t] = s; __syncthreads();
  for (int o = 64; o > 0; o >>= 1) { if (t < o) red[t] += red[t + o]; __syncthreads(); }
  float mean = red[0] / (float)C;
  __syncthreads();
  float vs = 0.f;
  for (int i = 0; i < per; ++i) { float d = loc[i] - mean; vs += d * d; }
  red[t] = vs; __syncthreads();
  for (int o = 64; o > 0; o >>= 1) { if (t < o) red[t] += red[t + o]; __syncthreads(); }
  float inv = rsqrtf(red[0] / (float)C + 1e-5f);
  float* yr = y + (size_t)row * C;
  for (int i = 0; i < per; ++i) {
    int c = t + (i << 7);
    yr[c] = (loc[i] - mean) * inv * g[c] + b[c];
  }
}

// ---------------------------------------------------------------------------
// Elementwise f32 -> bf16
// ---------------------------------------------------------------------------
__global__ void cvt_f32_bf16(const float* __restrict__ in, u16* __restrict__ out, int n) {
  int i = blockIdx.x * 256 + threadIdx.x;
  if (i < n) out[i] = f2bf(in[i]);
}

// Weight f32 [K,N] row-major -> bf16 [N,K] row-major (B^T for WMMA B-fragments)
__global__ void cvt_w_transpose(const float* __restrict__ in, u16* __restrict__ out,
                                int K, int N) {
  int i = blockIdx.x * 256 + threadIdx.x;
  if (i >= K * N) return;
  int k = i / N, n = i - k * N;
  out[(size_t)n * K + k] = f2bf(in[i]);
}

// ---------------------------------------------------------------------------
// bf16 WMMA GEMM: out[M,N] = A[M,K](bf16) * Bt[N,K]^T(bf16) + bias (+GELU)(+resid)
// Block: 256 threads (8 waves), tile 128x128, BK=32. Wave -> 32x64 (2x4 wmma).
// Double-buffered LDS; full tiles staged with global_load_async_to_lds_b128 so
// the async DMA of tile k+1 overlaps the WMMAs of tile k.
// ---------------------------------------------------------------------------
__global__ __launch_bounds__(256)
void gemm_bf16_wmma(const u16* __restrict__ A, const u16* __restrict__ Bt,
                    const float* __restrict__ bias, const float* __restrict__ resid,
                    float* __restrict__ out, int M, int N, int K, int act_gelu) {
  __shared__ __align__(16) u16 sA[2][128 * 32];
  __shared__ __align__(16) u16 sB[2][128 * 32];
  const int t = threadIdx.x;
  const int mBase = blockIdx.y * 128;
  const int nBase = blockIdx.x * 128;
  const int lane = t & 31, wave = t >> 5;
  const int wm = (wave >> 1) * 32;   // wave M offset: 0/32/64/96
  const int wn = (wave & 1) * 64;    // wave N offset: 0/64
  const int r16 = lane & 15, hi = lane >> 4;
  const bool full = (mBase + 128 <= M) && (nBase + 128 <= N);  // uniform per block

  v8f acc[2][4];
  {
    v8f z = {};
    for (int s = 0; s < 2; ++s)
      for (int j = 0; j < 4; ++j) acc[s][j] = z;
  }

  // Stage one 128x32 bf16 tile of A and Bt into LDS buffer `buf`.
  // Full tiles: per-lane async DMA straight into LDS (no VGPR round trip).
  // Partial tiles (conv tail blocks): guarded loads + ds_store, zero fill.
  auto issue = [&](int buf, int k0) {
    if (full) {
#pragma unroll
      for (int c = t; c < 512; c += 256) {
        int row = c >> 2, co = (c & 3) * 8;                 // 8 bf16 = 16B chunk
        const u16* gA = A + (size_t)(mBase + row) * K + k0 + co;
        u32 lA = (u32)(size_t)&sA[buf][row * 32 + co];
        asm volatile("global_load_async_to_lds_b128 %0, %1, off"
                     :: "v"(lA), "v"(gA) : "memory");
        const u16* gB = Bt + (size_t)(nBase + row) * K + k0 + co;
        u32 lB = (u32)(size_t)&sB[buf][row * 32 + co];
        asm volatile("global_load_async_to_lds_b128 %0, %1, off"
                     :: "v"(lB), "v"(gB) : "memory");
      }
    } else {
#pragma unroll
      for (int c = t; c < 512; c += 256) {
        int row = c >> 2, co = (c & 3) * 8;
        v4u va = {0, 0, 0, 0};
        if (mBase + row < M) va = *(const v4u*)(A + (size_t)(mBase + row) * K + k0 + co);
        *(v4u*)&sA[buf][row * 32 + co] = va;
        v4u vb = {0, 0, 0, 0};
        if (nBase + row < N) vb = *(const v4u*)(Bt + (size_t)(nBase + row) * K + k0 + co);
        *(v4u*)&sB[buf][row * 32 + co] = vb;
      }
    }
  };

  const int nk = K >> 5;  // K is a multiple of 32 for every GEMM here
  issue(0, 0);
  for (int ki = 0; ki < nk; ++ki) {
    const int cur = ki & 1;
    // Own async copies for tile `cur` done, then workgroup-wide visibility.
    asm volatile("s_wait_asynccnt 0x0" ::: "memory");
    __syncthreads();
    // Prefetch next tile into the other buffer; overlaps the WMMAs below.
    if (ki + 1 < nk) issue(cur ^ 1, (ki + 1) << 5);

    Frag16 a[2], b[4];
    // A 16x32 bf16 layout: lanes 0-15 row M=lane hold K kb..kb+7 (v0-3) and
    // kb+16..kb+23 (v4-7) with kb = (lane>>4)*8.
#pragma unroll
    for (int s = 0; s < 2; ++s) {
      const u16* p = &sA[cur][(wm + s * 16 + r16) * 32 + hi * 8];
      a[s].u[0] = *(const v4u*)p;
      a[s].u[1] = *(const v4u*)(p + 16);
    }
    // B 32x16 bf16 layout: col N=lane%16 holds K kb..kb+7 (v0-3) and
    // kb+16..kb+23 (v4-7) with kb = (lane>>4)*8 (same as A).
#pragma unroll
    for (int j = 0; j < 4; ++j) {
      const u16* p = &sB[cur][(wn + j * 16 + r16) * 32 + hi * 8];
      b[j].u[0] = *(const v4u*)p;
      b[j].u[1] = *(const v4u*)(p + 16);
    }
#pragma unroll
    for (int s = 0; s < 2; ++s)
#pragma unroll
      for (int j = 0; j < 4; ++j)
        acc[s][j] = __builtin_amdgcn_wmma_f32_16x16x32_bf16(
            false, a[s].v, false, b[j].v, (short)0, acc[s][j], false, false);
  }

  // Epilogue. C/D layout: lanes 0-15 -> N=lane, M=vgpr(0..7); lanes 16-31 -> M=8+vgpr.
#pragma unroll
  for (int s = 0; s < 2; ++s)
#pragma unroll
    for (int j = 0; j < 4; ++j) {
      int n = nBase + wn + j * 16 + r16;
      if (n >= N) continue;
      float bv = bias ? bias[n] : 0.f;
#pragma unroll
      for (int rr = 0; rr < 8; ++rr) {
        int m = mBase + wm + s * 16 + hi * 8 + rr;
        if (m >= M) continue;
        float v = acc[s][j][rr] + bv;
        if (act_gelu) v = 0.5f * v * (1.f + erff(v * 0.70710678118654752f));
        if (resid) v += resid[(size_t)m * N + n];
        out[(size_t)m * N + n] = v;
      }
    }
}

// ---------------------------------------------------------------------------
// Neighborhood attention, KS=7, HEADS=4, d=32, 56x56 grid.
// One thread per (pixel, head); online softmax over the 49 neighbors.
// ---------------------------------------------------------------------------
__global__ __launch_bounds__(128)
void natten_kernel(const float* __restrict__ Q, int qs,
                   const float* __restrict__ Kp, const float* __restrict__ Vp, int ks,
                   const float* __restrict__ rpb, float* __restrict__ out, int Mtot) {
  int idx = blockIdx.x * 128 + threadIdx.x;
  int head = idx & 3;
  int m = idx >> 2;
  if (m >= Mtot) return;
  int xw = m % 56;
  int y = (m / 56) % 56;
  int bb = m / 3136;

  const float* qr = Q + (size_t)m * qs + head * 32;
  float q[32];
#pragma unroll
  for (int d = 0; d < 32; ++d) q[d] = qr[d] * 0.17677669529663687f;  // 32^-0.5

  int sy = y - 3;  sy = sy < 0 ? 0 : (sy > 49 ? 49 : sy);
  int sx = xw - 3; sx = sx < 0 ? 0 : (sx > 49 ? 49 : sx);
  const float* rp = rpb + head * 169;

  float mx = -3.4e38f, l = 0.f;
  float acc[32];
#pragma unroll
  for (int d = 0; d < 32; ++d) acc[d] = 0.f;

  for (int i = 0; i < 7; ++i) {
    int ny = sy + i, ri = ny - y + 6;
    for (int j = 0; j < 7; ++j) {
      int nx = sx + j, rj = nx - xw + 6;
      int nm = bb * 3136 + ny * 56 + nx;
      const float* kr = Kp + (size_t)nm * ks + head * 32;
      const float* vr = Vp + (size_t)nm * ks + head * 32;
      float sc = rp[ri * 13 + rj];
#pragma unroll
      for (int d = 0; d < 32; ++d) sc += q[d] * kr[d];
      float mnew = fmaxf(mx, sc);
      float corr = __expf(mx - mnew);
      float p = __expf(sc - mnew);
      l = l * corr + p;
#pragma unroll
      for (int d = 0; d < 32; ++d) acc[d] = acc[d] * corr + p * vr[d];
      mx = mnew;
    }
  }
  float invl = 1.f / l;
  float* orow = out + (size_t)m * 128 + head * 32;
#pragma unroll
  for (int d = 0; d < 32; ++d) orow[d] = acc[d] * invl;
}

// ---------------------------------------------------------------------------
// im2col for 3x3 stride-2 pad-1 conv over concat([x, x_cross]) (256 ch) -> bf16
// out: [3136, 2304], k = (ky*3+kx)*256 + c  (matches HWIO weight flattening)
// ---------------------------------------------------------------------------
__global__ void im2col_kernel(const float* __restrict__ xa, const float* __restrict__ xb,
                              u16* __restrict__ out) {
  int i = blockIdx.x * 256 + threadIdx.x;
  const int TOT = 3136 * 2304;
  if (i >= TOT) return;
  int m = i / 2304, k = i - m * 2304;
  int ky = k / 768, rem = k - ky * 768;
  int kx = rem / 256, c = rem - kx * 256;
  int bb = m / 784, p = m - bb * 784;
  int oy = p / 28, ox = p - oy * 28;
  int iy = oy * 2 + ky - 1, ix = ox * 2 + kx - 1;
  float v = 0.f;
  if (iy >= 0 && iy < 56 && ix >= 0 && ix < 56) {
    size_t off = (((size_t)bb * 56 + iy) * 56 + ix) * 128 + (c & 127);
    v = (c < 128) ? xa[off] : xb[off];
  }
  out[i] = f2bf(v);
}

// ---------------------------------------------------------------------------
// Host orchestration
// ---------------------------------------------------------------------------
extern "C" void kernel_launch(void* const* d_in, const int* in_sizes, int n_in,
                              void* d_out, int out_size, void* d_ws, size_t ws_size,
                              hipStream_t stream) {
  (void)in_sizes; (void)n_in; (void)out_size; (void)ws_size;
  const int M = 4 * 56 * 56;  // 12544 pixels

  const float* x        = (const float*)d_in[0];
  const float* x_multi  = (const float*)d_in[1];
  const float* sa_n1g   = (const float*)d_in[2];
  const float* sa_n1b   = (const float*)d_in[3];
  const float* sa_qkv_w = (const float*)d_in[4];
  const float* sa_qkv_b = (const float*)d_in[5];
  const float* sa_rpb   = (const float*)d_in[6];
  const float* sa_pw    = (const float*)d_in[7];
  const float* sa_pb    = (const float*)d_in[8];
  const float* sa_n2g   = (const float*)d_in[9];
  const float* sa_n2b   = (const float*)d_in[10];
  const float* sa_f1w   = (const float*)d_in[11];
  const float* sa_f1b   = (const float*)d_in[12];
  const float* sa_f2w   = (const float*)d_in[13];
  const float* sa_f2b   = (const float*)d_in[14];
  const float* cr_n1g   = (const float*)d_in[15];
  const float* cr_n1b   = (const float*)d_in[16];
  const float* cr_qw    = (const float*)d_in[17];
  const float* cr_qb    = (const float*)d_in[18];
  const float* cr_kvw   = (const float*)d_in[19];
  const float* cr_kvb   = (const float*)d_in[20];
  const float* cr_rpb   = (const float*)d_in[21];
  const float* cr_pw    = (const float*)d_in[22];
  const float* cr_pb    = (const float*)d_in[23];
  const float* cr_n2g   = (const float*)d_in[24];
  const float* cr_n2b   = (const float*)d_in[25];
  const float* cr_f1w   = (const float*)d_in[26];
  const float* cr_f1b   = (const float*)d_in[27];
  const float* cr_f2w   = (const float*)d_in[28];
  const float* cr_f2b   = (const float*)d_in[29];
  const float* ds_w     = (const float*)d_in[30];
  const float* ds_ng    = (const float*)d_in[31];
  const float* ds_nb    = (const float*)d_in[32];

  // workspace arena
  char* ws = (char*)d_ws;
  const size_t SZ_MC  = (size_t)M * 128 * 4;   // 6.42 MB
  const size_t SZ_BIG = (size_t)M * 512 * 4;   // 25.7 MB
  const size_t SZ_ABF = (size_t)M * 512 * 2;   // 12.8 MB
  float* xa    = (float*)(ws);
  float* xb    = (float*)(ws + SZ_MC);
  float* lnb   = (float*)(ws + 2 * SZ_MC);
  float* attn  = (float*)(ws + 3 * SZ_MC);
  float* big   = (float*)(ws + 4 * SZ_MC);
  u16*   abf   = (u16*)  (ws + 4 * SZ_MC + SZ_BIG);
  u16*   wbf   = (u16*)  (ws + 4 * SZ_MC + SZ_BIG + SZ_ABF);
  u16*   imcol = (u16*)big;     // conv time: big free (<= SZ_BIG)
  float* convo = attn;          // conv time: attn free (same size)

  hipMemcpyAsync(xa, x, SZ_MC, hipMemcpyDeviceToDevice, stream);
  hipMemcpyAsync(xb, x, SZ_MC, hipMemcpyDeviceToDevice, stream);

  auto LN = [&](const float* s, const float* g, const float* b, float* d, int rows, int C) {
    ln_kernel<<<rows, 128, 0, stream>>>(s, g, b, d, C);
  };
  auto CVT = [&](const float* s, u16* d, int n) {
    cvt_f32_bf16<<<(n + 255) / 256, 256, 0, stream>>>(s, d, n);
  };
  auto CVTW = [&](const float* s, int K, int N) {
    cvt_w_transpose<<<(K * N + 255) / 256, 256, 0, stream>>>(s, wbf, K, N);
  };
  auto GEMM = [&](const u16* A, const float* bias, const float* resid, float* o,
                  int Mm, int Nn, int Kk, int gelu) {
    dim3 g((Nn + 127) / 128, (Mm + 127) / 128);
    gemm_bf16_wmma<<<g, 256, 0, stream>>>(A, wbf, bias, resid, o, Mm, Nn, Kk, gelu);
  };
  auto NA = [&](const float* Q, int qs, const float* Kp, const float* Vp, int ks,
                const float* rpb) {
    natten_kernel<<<(M * 4 + 127) / 128, 128, 0, stream>>>(Q, qs, Kp, Vp, ks, rpb, attn, M);
  };

  // ---- self-attention NAT layers on xa ----
  for (int i = 0; i < 2; ++i) {
    LN(xa, sa_n1g + i * 128, sa_n1b + i * 128, lnb, M, 128);
    CVT(lnb, abf, M * 128);
    CVTW(sa_qkv_w + (size_t)i * 128 * 384, 128, 384);
    GEMM(abf, sa_qkv_b + i * 384, nullptr, big, M, 384, 128, 0);        // qkv [M,384]
    NA(big, 384, big + 128, big + 256, 384, sa_rpb + (size_t)i * 676);  // attn [M,128]
    CVT(attn, abf, M * 128);
    CVTW(sa_pw + (size_t)i * 128 * 128, 128, 128);
    GEMM(abf, sa_pb + i * 128, xa, xa, M, 128, 128, 0);                 // x += proj
    LN(xa, sa_n2g + i * 128, sa_n2b + i * 128, lnb, M, 128);
    CVT(lnb, abf, M * 128);
    CVTW(sa_f1w + (size_t)i * 128 * 512, 128, 512);
    GEMM(abf, sa_f1b + i * 512, nullptr, big, M, 512, 128, 1);          // gelu(fc1)
    CVT(big, abf, M * 512);
    CVTW(sa_f2w + (size_t)i * 512 * 128, 512, 128);
    GEMM(abf, sa_f2b + i * 128, xa, xa, M, 128, 512, 0);                // x += fc2
  }

  // ---- cross NAT layers: q from xb chain, kv from LN(x_multi) ----
  float* qbuf = big + (size_t)M * 256;  // [M,128] after kv region
  for (int i = 0; i < 2; ++i) {
    CVT(xb, abf, M * 128);
    CVTW(cr_qw + (size_t)i * 128 * 128, 128, 128);
    GEMM(abf, cr_qb + i * 128, nullptr, qbuf, M, 128, 128, 0);          // q [M,128]
    LN(x_multi, cr_n1g + i * 128, cr_n1b + i * 128, lnb, M, 128);
    CVT(lnb, abf, M * 128);
    CVTW(cr_kvw + (size_t)i * 128 * 256, 128, 256);
    GEMM(abf, cr_kvb + i * 256, nullptr, big, M, 256, 128, 0);          // kv [M,256]
    NA(qbuf, 128, big, big + 128, 256, cr_rpb + (size_t)i * 676);
    CVT(attn, abf, M * 128);
    CVTW(cr_pw + (size_t)i * 128 * 128, 128, 128);
    GEMM(abf, cr_pb + i * 128, x_multi, xb, M, 128, 128, 0);            // xb = x_multi + proj
    LN(xb, cr_n2g + i * 128, cr_n2b + i * 128, lnb, M, 128);
    CVT(lnb, abf, M * 128);
    CVTW(cr_f1w + (size_t)i * 128 * 512, 128, 512);
    GEMM(abf, cr_f1b + i * 512, nullptr, big, M, 512, 128, 1);
    CVT(big, abf, M * 512);
    CVTW(cr_f2w + (size_t)i * 512 * 128, 512, 128);
    GEMM(abf, cr_f2b + i * 128, xb, xb, M, 128, 512, 0);
  }

  // ---- downsample: concat -> 3x3/s2 conv (implicit GEMM) -> LayerNorm ----
  im2col_kernel<<<(3136 * 2304 + 255) / 256, 256, 0, stream>>>(xa, xb, imcol);
  CVTW(ds_w, 2304, 512);                          // HWIO flat == [K=2304, N=512]
  GEMM(imcol, nullptr, nullptr, convo, 3136, 512, 2304, 0);
  LN(convo, ds_ng, ds_nb, (float*)d_out, 3136, 512);
}
